// SpatialMambaBlock_14688788152945
// MI455X (gfx1250) — compile-verified, matrix operand order checked by reading, not run on hardware
//
#include <hip/hip_runtime.h>
#include <hip/hip_bf16.h>
#include <math.h>

// ---- problem dimensions (compile-time, from reference) ----
#define D_MODEL 1024
#define D_STATE 16
#define D_CONV  4
#define D_INNER 2048            // D_MODEL * EXPAND
#define DT_RANK 64              // (D_MODEL + 15) / 16
#define BATCH   4
#define SEQ     2048
#define BL      (BATCH * SEQ)   // 8192 rows for all GEMMs
#define DBL_W   (DT_RANK + 2 * D_STATE)  // 96 real columns
#define DBL_PAD 128             // padded to a multiple of the 64-wide N tile

typedef __attribute__((ext_vector_type(16))) __bf16 v16bf;
typedef __attribute__((ext_vector_type(8)))  __bf16 v8bf;
typedef __attribute__((ext_vector_type(8)))  float  v8f;

__device__ __forceinline__ float sigmoidf_(float x) {
    return 1.0f / (1.0f + __expf(-x));
}

// ---------------------------------------------------------------------------
// elementwise f32 -> bf16 cast
// ---------------------------------------------------------------------------
__global__ void cast_f32_to_bf16(const float* __restrict__ in,
                                 __bf16* __restrict__ out, long n) {
    long i = (long)blockIdx.x * blockDim.x + threadIdx.x;
    if (i < n) out[i] = (__bf16)in[i];
}

// W_x (96 x 2048) -> bf16, zero-padded to 128 x 2048
__global__ void cast_pad_wx_bf16(const float* __restrict__ in,
                                 __bf16* __restrict__ out) {
    long i = (long)blockIdx.x * blockDim.x + threadIdx.x;
    if (i >= (long)DBL_PAD * D_INNER) return;
    long r = i / D_INNER;
    long c = i - r * D_INNER;
    out[i] = (r < DBL_W) ? (__bf16)in[r * D_INNER + c] : (__bf16)0.0f;
}

// extract dbl[:, :DT_RANK] (row stride DBL_PAD) into contiguous bf16 (stride 64)
__global__ void extract_dt_bf16(const float* __restrict__ dbl,
                                __bf16* __restrict__ out) {
    long i = (long)blockIdx.x * blockDim.x + threadIdx.x;
    if (i >= (long)BL * DT_RANK) return;
    long r = i / DT_RANK;
    long c = i - r * DT_RANK;
    out[i] = (__bf16)dbl[r * DBL_PAD + c];
}

// ---------------------------------------------------------------------------
// depthwise causal conv (k=4) + SiLU, reading x_in = xz[..., :D_INNER]
// writes f32 (for the scan) and bf16 (for the W_x GEMM)
// ---------------------------------------------------------------------------
__global__ void conv_silu_kernel(const float* __restrict__ xz,
                                 const float* __restrict__ convw,
                                 float* __restrict__ xcf,
                                 __bf16* __restrict__ xcb) {
    long i = (long)blockIdx.x * blockDim.x + threadIdx.x;
    if (i >= (long)BL * D_INNER) return;
    int  d    = (int)(i % D_INNER);
    long bl   = i / D_INNER;
    int  l    = (int)(bl % SEQ);
    long brow = bl - l;                 // b * SEQ
    float s = 0.0f;
#pragma unroll
    for (int t = 0; t < D_CONV; ++t) {
        int ls = l - (D_CONV - 1) + t;
        if (ls >= 0)
            s += convw[d * D_CONV + t] * xz[(brow + ls) * (2L * D_INNER) + d];
    }
    s = s * sigmoidf_(s);               // SiLU
    xcf[i] = s;
    xcb[i] = (__bf16)s;
}

// ---------------------------------------------------------------------------
// bf16 WMMA GEMM: C[m][n] = sum_k A[m][k] * Bw[n][k]   (i.e. A @ Bw^T)
// block = 256 threads = 8 waves, arranged 4(M) x 2(N); block tile 128M x 64N.
// Each wave owns a 32x32 sub-tile = 2x2 WMMA accumulators, so the k-loop
// issues 8 global_load_b128 per 4 v_wmma (2 loads/WMMA, A/B each reused 2x).
// Requirements: M % 128 == 0, N % 64 == 0, K % 32 == 0 (true for all calls).
// ---------------------------------------------------------------------------
__global__ __launch_bounds__(256)
void gemm_bf16_wmma(const __bf16* __restrict__ A,  long lda,
                    const __bf16* __restrict__ Bw, long ldb,
                    float* __restrict__ C,         long ldc,
                    long K) {
    const int wave  = threadIdx.x >> 5;
    const int waveM = wave >> 1;       // 0..3
    const int waveN = wave & 1;        // 0..1
    const int lane  = threadIdx.x & 31;
    const int row   = lane & 15;       // M-row (A) / N-col (B) within tile
    const int kh    = lane >> 4;       // which K-half this lane holds

    const long m0 = (long)blockIdx.y * 128 + (long)waveM * 32;
    const long n0 = (long)blockIdx.x * 64 + (long)waveN * 32;

    v8f acc00 = {}, acc01 = {}, acc10 = {}, acc11 = {};

    // ISA bf16 A/B layout: lanes 0-15 hold K 0..7 (vgpr0-3) and 16..23
    // (vgpr4-7); lanes 16-31 hold K 8..15 and 24..31.
    const __bf16* apm0 = A  + (m0 + row)      * lda + kh * 8;
    const __bf16* apm1 = A  + (m0 + 16 + row) * lda + kh * 8;
    const __bf16* bpn0 = Bw + (n0 + row)      * ldb + kh * 8;
    const __bf16* bpn1 = Bw + (n0 + 16 + row) * ldb + kh * 8;

    for (long k0 = 0; k0 < K; k0 += 32) {
        v8bf a0lo = *(const v8bf*)(apm0 + k0);
        v8bf a0hi = *(const v8bf*)(apm0 + k0 + 16);
        v8bf a1lo = *(const v8bf*)(apm1 + k0);
        v8bf a1hi = *(const v8bf*)(apm1 + k0 + 16);
        v8bf b0lo = *(const v8bf*)(bpn0 + k0);
        v8bf b0hi = *(const v8bf*)(bpn0 + k0 + 16);
        v8bf b1lo = *(const v8bf*)(bpn1 + k0);
        v8bf b1hi = *(const v8bf*)(bpn1 + k0 + 16);

        v16bf a0, a1, b0, b1;
#pragma unroll
        for (int e = 0; e < 8; ++e) {
            a0[e] = a0lo[e]; a0[e + 8] = a0hi[e];
            a1[e] = a1lo[e]; a1[e + 8] = a1hi[e];
            b0[e] = b0lo[e]; b0[e + 8] = b0hi[e];
            b1[e] = b1lo[e]; b1[e + 8] = b1hi[e];
        }
        acc00 = __builtin_amdgcn_wmma_f32_16x16x32_bf16(
                    false, a0, false, b0, (short)0, acc00, false, false);
        acc01 = __builtin_amdgcn_wmma_f32_16x16x32_bf16(
                    false, a0, false, b1, (short)0, acc01, false, false);
        acc10 = __builtin_amdgcn_wmma_f32_16x16x32_bf16(
                    false, a1, false, b0, (short)0, acc10, false, false);
        acc11 = __builtin_amdgcn_wmma_f32_16x16x32_bf16(
                    false, a1, false, b1, (short)0, acc11, false, false);
    }

    // C/D layout: lanes 0-15 -> N=lane, M=r; lanes 16-31 -> N=lane-16, M=r+8
#pragma unroll
    for (int r = 0; r < 8; ++r) {
        long mA = m0 + r + kh * 8;
        long mB = mA + 16;
        C[mA * ldc + n0 + row]      = acc00[r];
        C[mA * ldc + n0 + 16 + row] = acc01[r];
        C[mB * ldc + n0 + row]      = acc10[r];
        C[mB * ldc + n0 + 16 + row] = acc11[r];
    }
}

// ---------------------------------------------------------------------------
// selective scan: one thread per (batch, channel); h[16] in registers.
// b is derived from blockIdx so the B/C row pointer into dbl is provably
// wave-uniform (scalar loads). Prefetch per-lane streams 16 steps ahead.
// Fuses: softplus(dt_raw + b_dt), recurrence, y = h.C, + Dp*x_conv,
//        * SiLU(z), cast to bf16 for the output GEMM.
// Launch: grid = BATCH * (D_INNER/256) blocks of 256 threads.
// ---------------------------------------------------------------------------
#define BLK_PER_B (D_INNER / 256)   // 8
__global__ __launch_bounds__(256)
void mamba_scan(const float* __restrict__ dtraw,
                const float* __restrict__ b_dt,
                const float* __restrict__ xconv,
                const float* __restrict__ dbl,
                const float* __restrict__ xz,
                const float* __restrict__ A_log,
                const float* __restrict__ Dp,
                __bf16* __restrict__ ybf) {
    const int b = blockIdx.x / BLK_PER_B;                    // uniform
    const int d = (blockIdx.x % BLK_PER_B) * 256 + threadIdx.x;

    float a[D_STATE], h[D_STATE];
#pragma unroll
    for (int n = 0; n < D_STATE; ++n) {
        a[n] = -__expf(A_log[d * D_STATE + n]);   // A = -exp(A_log)
        h[n] = 0.0f;
    }
    const float bdt = b_dt[d];
    const float dpv = Dp[d];

    const long base = (long)b * SEQ;
    for (int l = 0; l < SEQ; ++l) {
        long bl = base + l;
        if (l + 16 < SEQ) {            // uniform branch: prefetch lane streams
            long pf = bl + 16;
            __builtin_prefetch(&dtraw[pf * D_INNER + d], 0, 1);
            __builtin_prefetch(&xconv[pf * D_INNER + d], 0, 1);
            __builtin_prefetch(&xz[pf * (2L * D_INNER) + D_INNER + d], 0, 1);
        }
        float dtv = dtraw[bl * D_INNER + d] + bdt;
        dtv = (dtv > 20.0f) ? dtv : log1pf(__expf(dtv));   // softplus
        float xv = xconv[bl * D_INNER + d];
        const float* bc = dbl + bl * DBL_PAD;              // uniform -> s_load
        float y = 0.0f;
#pragma unroll
        for (int n = 0; n < D_STATE; ++n) {
            float Bn = bc[DT_RANK + n];
            float Cn = bc[DT_RANK + D_STATE + n];
            h[n] = h[n] * __expf(dtv * a[n]) + dtv * Bn * xv;
            y   += h[n] * Cn;
        }
        float zv  = xz[bl * (2L * D_INNER) + D_INNER + d];
        float out = (y + dpv * xv) * (zv * sigmoidf_(zv)); // * SiLU(z)
        ybf[bl * D_INNER + d] = (__bf16)out;
    }
}

// ---------------------------------------------------------------------------
extern "C" void kernel_launch(void* const* d_in, const int* in_sizes, int n_in,
                              void* d_out, int out_size, void* d_ws, size_t ws_size,
                              hipStream_t stream) {
    const float* x     = (const float*)d_in[0];
    const float* W_in  = (const float*)d_in[1];
    const float* convw = (const float*)d_in[2];
    const float* W_x   = (const float*)d_in[3];
    const float* W_dt  = (const float*)d_in[4];
    const float* b_dt  = (const float*)d_in[5];
    const float* A_log = (const float*)d_in[6];
    const float* Dp    = (const float*)d_in[7];
    const float* W_out = (const float*)d_in[8];
    float* out = (float*)d_out;

    // bump allocator over d_ws
    char* p = (char*)d_ws;
    auto alloc = [&](size_t bytes) -> char* {
        char* r = p;
        p += (bytes + 255) & ~(size_t)255;
        return r;
    };
    float*  xz       = (float*) alloc(sizeof(float)  * (size_t)BL * 2 * D_INNER); // 134 MB
    float*  xconv_f  = (float*) alloc(sizeof(float)  * (size_t)BL * D_INNER);     //  67 MB
    __bf16* xconv_b  = (__bf16*)alloc(sizeof(__bf16) * (size_t)BL * D_INNER);     //  34 MB
    float*  dbl      = (float*) alloc(sizeof(float)  * (size_t)BL * DBL_PAD);     //   4 MB
    __bf16* dblr_b   = (__bf16*)alloc(sizeof(__bf16) * (size_t)BL * DT_RANK);     //   1 MB
    float*  dtraw    = (float*) alloc(sizeof(float)  * (size_t)BL * D_INNER);     //  67 MB
    __bf16* y_b      = (__bf16*)alloc(sizeof(__bf16) * (size_t)BL * D_INNER);     //  34 MB
    __bf16* x_b      = (__bf16*)alloc(sizeof(__bf16) * (size_t)BL * D_MODEL);     //  17 MB
    __bf16* Win_b    = (__bf16*)alloc(sizeof(__bf16) * (size_t)2 * D_INNER * D_MODEL);
    __bf16* Wx_b     = (__bf16*)alloc(sizeof(__bf16) * (size_t)DBL_PAD * D_INNER);
    __bf16* Wdt_b    = (__bf16*)alloc(sizeof(__bf16) * (size_t)D_INNER * DT_RANK);
    __bf16* Wout_b   = (__bf16*)alloc(sizeof(__bf16) * (size_t)D_MODEL * D_INNER);
    (void)ws_size; (void)in_sizes; (void)n_in; (void)out_size;

    const int T = 256;
    auto blocks = [](long n) { return dim3((unsigned)((n + 255) / 256)); };

    // 1) cast inputs / weights to bf16 (W_x zero-padded to 128 rows)
    cast_f32_to_bf16<<<blocks((long)BL * D_MODEL), T, 0, stream>>>(x, x_b, (long)BL * D_MODEL);
    cast_f32_to_bf16<<<blocks((long)2 * D_INNER * D_MODEL), T, 0, stream>>>(W_in, Win_b, (long)2 * D_INNER * D_MODEL);
    cast_pad_wx_bf16<<<blocks((long)DBL_PAD * D_INNER), T, 0, stream>>>(W_x, Wx_b);
    cast_f32_to_bf16<<<blocks((long)D_INNER * DT_RANK), T, 0, stream>>>(W_dt, Wdt_b, (long)D_INNER * DT_RANK);
    cast_f32_to_bf16<<<blocks((long)D_MODEL * D_INNER), T, 0, stream>>>(W_out, Wout_b, (long)D_MODEL * D_INNER);

    // 2) xz = x @ W_in^T        (8192 x 1024) x (4096 x 1024)^T
    gemm_bf16_wmma<<<dim3(2 * D_INNER / 64, BL / 128), 256, 0, stream>>>(
        x_b, D_MODEL, Win_b, D_MODEL, xz, 2 * D_INNER, D_MODEL);

    // 3) depthwise causal conv + SiLU on x_in = xz[..., :D_INNER]
    conv_silu_kernel<<<blocks((long)BL * D_INNER), T, 0, stream>>>(xz, convw, xconv_f, xconv_b);

    // 4) dbl = x_conv @ W_x^T   (8192 x 2048) x (128 x 2048)^T  [padded N]
    gemm_bf16_wmma<<<dim3(DBL_PAD / 64, BL / 128), 256, 0, stream>>>(
        xconv_b, D_INNER, Wx_b, D_INNER, dbl, DBL_PAD, D_INNER);

    // 5) dt_raw = dbl[:, :64] @ W_dt^T   (8192 x 64) x (2048 x 64)^T
    extract_dt_bf16<<<blocks((long)BL * DT_RANK), T, 0, stream>>>(dbl, dblr_b);
    gemm_bf16_wmma<<<dim3(D_INNER / 64, BL / 128), 256, 0, stream>>>(
        dblr_b, DT_RANK, Wdt_b, DT_RANK, dtraw, D_INNER, DT_RANK);

    // 6) selective scan + gating, fused, -> y (bf16)
    mamba_scan<<<dim3(BATCH * BLK_PER_B), 256, 0, stream>>>(
        dtraw, b_dt, xconv_f, dbl, xz, A_log, Dp, y_b);

    // 7) out = y @ W_out^T      (8192 x 2048) x (1024 x 2048)^T
    gemm_bf16_wmma<<<dim3(D_MODEL / 64, BL / 128), 256, 0, stream>>>(
        y_b, D_INNER, Wout_b, D_INNER, out, D_MODEL, D_INNER);
}
